// Attention_LSTM_Decoder_42099269435777
// MI455X (gfx1250) — matrix operand order repaired and checked by reading, not compiled, not run on hardware
//
#include <hip/hip_runtime.h>
#include <hip/hip_bf16.h>

typedef __attribute__((ext_vector_type(16))) _Float16 v16h;
typedef __attribute__((ext_vector_type(8)))  _Float16 v8h;
typedef __attribute__((ext_vector_type(8)))  float    v8f;

#define BB 256   // batch
#define SS 256   // seq len
#define EE 2     // encoder dim
#define DD 128   // decoder dim
#define GG 512   // 4*DD gates

// CDNA5 has a hardware v_tanh_f32 trans op; use it if the builtin is exposed.
__device__ __forceinline__ float hw_tanh(float x) {
#if __has_builtin(__builtin_amdgcn_tanhf)
    return __builtin_amdgcn_tanhf(x);
#elif __has_builtin(__builtin_amdgcn_tanh_f32)
    return __builtin_amdgcn_tanh_f32(x);
#else
    x = fminf(fmaxf(x, -15.0f), 15.0f);
    float e = __expf(2.0f * x);
    return (e - 1.0f) / (e + 1.0f);
#endif
}
__device__ __forceinline__ float hw_rcp(float x) {
#if __has_builtin(__builtin_amdgcn_rcpf)
    return __builtin_amdgcn_rcpf(x);
#else
    return 1.0f / x;
#endif
}
// sigmoid(x) = 0.5*tanh(x/2) + 0.5  -> single trans op
__device__ __forceinline__ float fast_sig(float x) {
    return 0.5f * hw_tanh(0.5f * x) + 0.5f;
}
__device__ __forceinline__ float wave_sum(float v) {
#pragma unroll
    for (int off = 16; off; off >>= 1) v += __shfl_xor(v, off, 32);
    return v;
}

// ---------------------------------------------------------------- precompute
__global__ void k_w1sum(const float* __restrict__ W1w, float* __restrict__ w1sum) {
    int s = threadIdx.x;
    const float* row = W1w + s * SS;
    float acc = 0.f;
#pragma unroll 8
    for (int j = 0; j < SS; ++j) acc += row[j];
    w1sum[s] = acc;
}

// a[b][s] = enc_flat[b] . W2_w[s] + W2_b[s] + W1_b[s]
__global__ void k_w2a(const float* __restrict__ enc, const float* __restrict__ W2w,
                      const float* __restrict__ W2b, const float* __restrict__ W1b,
                      float* __restrict__ aterm) {
    __shared__ float se[2 * SS];
    int b = blockIdx.x, s = threadIdx.x;
    se[s]      = enc[b * 2 * SS + s];
    se[s + SS] = enc[b * 2 * SS + SS + s];
    __builtin_prefetch(&W2w[s * 2 * SS], 0, 1);   // global_prefetch_b8
    __syncthreads();
    const float* row = W2w + s * 2 * SS;
    float acc = 0.f;
#pragma unroll 8
    for (int k = 0; k < 2 * SS; ++k) acc += se[k] * row[k];
    aterm[b * SS + s] = acc + W2b[s] + W1b[s];
}

// ---------------------------------------------------------------- main scan
// one workgroup per batch element; 256 threads = 8 wave32s; 256 local steps
__global__ void __launch_bounds__(256, 1)
k_decode(const float* __restrict__ enc, const float* __restrict__ aterm,
         const float* __restrict__ w1sum, const float* __restrict__ Wih,
         const float* __restrict__ Whh, const float* __restrict__ bih,
         const float* __restrict__ bhh, float* __restrict__ out) {
    __shared__ float sh_bias[GG], sh_wih[GG * EE];
    __shared__ float sh_alpha[8 * SS];           // per-wave softmax partials
    __shared__ float sh_red0[SS], sh_red1[SS];
    __shared__ float sh_gates[GG];
    __shared__ float sh_h[DD], sh_c[DD];
    __shared__ __align__(16) _Float16 sh_h16[DD];
    __shared__ float sh_ctx[2];

    const int b = blockIdx.x, tid = threadIdx.x;
    const int wave = tid >> 5, lane = tid & 31;
    const int n = lane & 15, hiK = lane >> 4;

    // ------- one-time LDS init (biases, W_ih, zeroed h/c state)
    sh_bias[tid]       = bih[tid]       + bhh[tid];
    sh_bias[tid + 256] = bih[tid + 256] + bhh[tid + 256];
#pragma unroll
    for (int q = 0; q < 4; ++q) sh_wih[tid + 256 * q] = Wih[tid + 256 * q];
    if (tid < DD) { sh_h[tid] = 0.f; sh_c[tid] = 0.f; sh_h16[tid] = (_Float16)0.f; }

    // ------- step-invariant per-lane data held in registers for all 256 steps
    float a8[8], m8[8];                 // lane's 8 softmax columns: s = lane + 32k
#pragma unroll
    for (int k = 0; k < 8; ++k) {
        const int s = lane + 32 * k;
        a8[k] = aterm[b * SS + s];
        m8[k] = w1sum[s];
    }
    const float e0 = enc[(b * SS + tid) * EE + 0];   // thread's encoder column
    const float e1 = enc[(b * SS + tid) * EE + 1];

    // ------- register-resident W_hh^T tiles (B operands), loop-invariant.
    // B layout (32x16 f16): lane n<16 = column n, K linear 0..15 (lanes 16-31: K 16..31)
    v16h Bt[4][4];
#pragma unroll
    for (int t4 = 0; t4 < 4; ++t4) {
        const int j0 = (wave * 4 + t4) * 16;
        const float* wr = Whh + (j0 + n) * DD;
#pragma unroll
        for (int kc = 0; kc < 4; ++kc) {
            const int kb = kc * 32 + hiK * 16;
            v16h bb;
#pragma unroll
            for (int i = 0; i < 16; ++i) bb[i] = (_Float16)wr[kb + i];
            Bt[t4][kc] = bb;
        }
    }
    __syncthreads();

    for (int step = 0; step < SS; ++step) {
        // ---- phase 1: en = tanh(a + h*m); per-d softmax; alpha partials in regs
        float acc8[8];
#pragma unroll
        for (int k = 0; k < 8; ++k) acc8[k] = 0.f;
#pragma unroll 2
        for (int r = 0; r < 16; ++r) {
            const float hval = sh_h[wave * 16 + r];
            float p[8], lsum = 0.f;
#pragma unroll
            for (int k = 0; k < 8; ++k) {
                float t = hw_tanh(a8[k] + hval * m8[k]);
                float e = __expf(t);    // t in (-1,1): no max-subtract needed
                p[k] = e; lsum += e;
            }
            const float inv = hw_rcp(wave_sum(lsum));
#pragma unroll
            for (int k = 0; k < 8; ++k) acc8[k] += p[k] * inv;
        }
#pragma unroll
        for (int k = 0; k < 8; ++k) sh_alpha[wave * SS + lane + 32 * k] = acc8[k];
        __syncthreads();

        // ---- phase 2: alpha (mean over d) and context = alpha . enc[b]
        {
            float acc = 0.f;
#pragma unroll
            for (int w = 0; w < 8; ++w) acc += sh_alpha[w * SS + tid];
            const float alpha = acc * (1.0f / (float)DD);
            sh_red0[tid] = alpha * e0;
            sh_red1[tid] = alpha * e1;
        }
        __syncthreads();
        if (wave < 2) {
            const float* r = (wave == 0) ? sh_red0 : sh_red1;
            float v = 0.f;
#pragma unroll
            for (int i = 0; i < 8; ++i) v += r[lane + 32 * i];
            v = wave_sum(v);
            if (lane == 0) sh_ctx[wave] = v;
        }
        __syncthreads();

        // ---- phase 3: gates_hh = h @ W_hh^T via WMMA (h broadcast to all A rows)
#pragma unroll
        for (int t4 = 0; t4 < 4; ++t4) {
            v8f c = {};
#pragma unroll
            for (int kc = 0; kc < 4; ++kc) {
                // A layout (16x32 f16): lanes 0-15 hold K kb..kb+15, lanes 16-31 K kb+16..kb+31
                const _Float16* hp = sh_h16 + kc * 32 + hiK * 16;
                v8h lo = *(const v8h*)(hp);
                v8h hi = *(const v8h*)(hp + 8);
                v16h A;
#pragma unroll
                for (int i = 0; i < 8; ++i) { A[i] = lo[i]; A[8 + i] = hi[i]; }
                c = __builtin_amdgcn_wmma_f32_16x16x32_f16(
                        false, A, false, Bt[t4][kc], (short)0, c, false, false);
            }
            if (lane < 16) sh_gates[(wave * 4 + t4) * 16 + lane] = c[0];  // row M=0
        }
        __syncthreads();

        // ---- phase 4: pointwise LSTM cell, write h_t
        if (tid < DD) {
            const int d = tid;
            const float x0 = sh_ctx[0], x1 = sh_ctx[1];
            float gi = sh_gates[d]           + x0 * sh_wih[2 * d]               + x1 * sh_wih[2 * d + 1]               + sh_bias[d];
            float gf = sh_gates[d + DD]      + x0 * sh_wih[2 * (d + DD)]        + x1 * sh_wih[2 * (d + DD) + 1]        + sh_bias[d + DD];
            float gg = sh_gates[d + 2 * DD]  + x0 * sh_wih[2 * (d + 2 * DD)]    + x1 * sh_wih[2 * (d + 2 * DD) + 1]    + sh_bias[d + 2 * DD];
            float go = sh_gates[d + 3 * DD]  + x0 * sh_wih[2 * (d + 3 * DD)]    + x1 * sh_wih[2 * (d + 3 * DD) + 1]    + sh_bias[d + 3 * DD];
            float c = fast_sig(gf) * sh_c[d] + fast_sig(gi) * hw_tanh(gg);
            float h = fast_sig(go) * hw_tanh(c);
            sh_c[d] = c; sh_h[d] = h; sh_h16[d] = (_Float16)h;
            out[((size_t)step * BB + b) * DD + d] = h;   // hs[S,B,D]
        }
        __syncthreads();
    }
}

extern "C" void kernel_launch(void* const* d_in, const int* in_sizes, int n_in,
                              void* d_out, int out_size, void* d_ws, size_t ws_size,
                              hipStream_t stream) {
    const float* enc = (const float*)d_in[0];
    const float* W1w = (const float*)d_in[1];
    const float* W1b = (const float*)d_in[2];
    const float* W2w = (const float*)d_in[3];
    const float* W2b = (const float*)d_in[4];
    const float* Wih = (const float*)d_in[5];
    const float* Whh = (const float*)d_in[6];
    const float* bih = (const float*)d_in[7];
    const float* bhh = (const float*)d_in[8];
    float* out = (float*)d_out;

    float* wsf   = (float*)d_ws;
    float* w1sum = wsf;          // 256 floats
    float* aterm = wsf + 256;    // 256*256 floats

    k_w1sum<<<1, 256, 0, stream>>>(W1w, w1sum);
    k_w2a<<<BB, 256, 0, stream>>>(enc, W2w, W2b, W1b, aterm);
    k_decode<<<BB, 256, 0, stream>>>(enc, aterm, w1sum, Wih, Whh, bih, bhh, out);
}
